// QKNorm_31731218382921
// MI455X (gfx1250) — compile-verified, matrix operand order checked by reading, not run on hardware
//
#include <hip/hip_runtime.h>
#include <hip/hip_bf16.h>

// ---------------------------------------------------------------------------
// Types for CDNA5 WMMA (wave32): v_wmma_f32_16x16x32_bf16
// ---------------------------------------------------------------------------
typedef __attribute__((ext_vector_type(16))) __bf16 bf16x16;
typedef __attribute__((ext_vector_type(2)))  __bf16 bf16x2;
typedef __attribute__((ext_vector_type(8)))  float  floatx8;

// Hardware f32 -> bf16 (RNE fptrunc; lowers to v_cvt_pk_bf16_f32)
__device__ __forceinline__ unsigned short bf16_bits(float f) {
  union { __bf16 b; unsigned short u; } c;
  c.b = (__bf16)f;
  return c.u;
}
__device__ __forceinline__ unsigned int pack_bf16x2(float lo, float hi) {
  bf16x2 p;
  p.x = (__bf16)lo;
  p.y = (__bf16)hi;
  union { bf16x2 v; unsigned int u; } c;
  c.v = p;
  return c.u;
}
__device__ __forceinline__ uint2 pack_bf16x4(float4 v) {
  uint2 r;
  r.x = pack_bf16x2(v.x, v.y);
  r.y = pack_bf16x2(v.z, v.w);
  return r;
}

// CDNA5 ISA 7.12.2, 16-bit 16x32 A/B fragment: a lane's 16 elements are the
// two contiguous 8-element runs row[hi*8 .. +7] and row[16+hi*8 .. +7].
// With K contiguous in LDS this is exactly two ds_load_b128.
union Frag { bf16x16 v; uint4 q[2]; };
__device__ __forceinline__ bf16x16 load_frag_row(const unsigned short* rowptr, int hi) {
  Frag f;
  f.q[0] = *(const uint4*)(rowptr + hi * 8);
  f.q[1] = *(const uint4*)(rowptr + 16 + hi * 8);
  return f.v;
}
__device__ __forceinline__ floatx8 wmma_bf16(bf16x16 a, bf16x16 b, floatx8 c) {
  return __builtin_amdgcn_wmma_f32_16x16x32_bf16(false, a, false, b, (short)0, c,
                                                 false, false);
}

// ---------------------------------------------------------------------------
// Kernel 1: qkv = X[M,K] @ W[K,N3]   (f32 in memory, bf16 WMMA, f32 accum)
// 128x128 tile per WG, 8 waves, each wave owns a 32x64 sub-tile (2x4 WMMAs).
// Double-buffered LDS software pipeline: next tile's global loads are in
// flight while the WMMAs run; one barrier per k-step.
// ---------------------------------------------------------------------------
__global__ __launch_bounds__(256) void qkv_gemm(const float* __restrict__ X,
                                                const float* __restrict__ W,
                                                float* __restrict__ C,
                                                int M, int N, int K) {
  __shared__ __align__(16) unsigned short sA[2][128][40];   // X tile  [m][k]
  __shared__ __align__(16) unsigned short sBt[2][128][40];  // W tile transposed [n][k]

  const int tid  = threadIdx.x;
  const int lane = tid & 31;
  const int wv   = tid >> 5;          // wave 0..7
  const int nl   = lane & 15;
  const int hi   = lane >> 4;
  const int wr   = wv & 3;            // 4 row groups of 32 rows
  const int wc   = wv >> 2;           // 2 col groups of 64 cols
  const int m0   = blockIdx.y * 128;
  const int n0   = blockIdx.x * 128;

  floatx8 acc[2][4];
  #pragma unroll
  for (int i = 0; i < 2; ++i)
    #pragma unroll
    for (int j = 0; j < 4; ++j) acc[i][j] = (floatx8){};

  float4 xv[4];
  float2 w0v[2], w1v[2];

  auto load_regs = [&](int kt) {
    #pragma unroll
    for (int ii = 0; ii < 4; ++ii) {
      int i = tid + ii * 256;               // 0..1023
      int r = i >> 3, c4 = (i & 7) << 2;    // 8 float4 per 32-wide row
      xv[ii] = *(const float4*)(X + (size_t)(m0 + r) * K + kt + c4);
    }
    #pragma unroll
    for (int ii = 0; ii < 2; ++ii) {
      int i = tid + ii * 256;               // 0..511
      int r2 = (i >> 6) << 1;               // k-row pair
      int c2 = (i & 63) << 1;               // col pair
      const float* wp = W + (size_t)(kt + r2) * N + n0 + c2;
      w0v[ii] = *(const float2*)wp;
      w1v[ii] = *(const float2*)(wp + N);
    }
  };
  auto store_lds = [&](int b) {
    #pragma unroll
    for (int ii = 0; ii < 4; ++ii) {
      int i = tid + ii * 256;
      int r = i >> 3, c4 = (i & 7) << 2;
      *(uint2*)&sA[b][r][c4] = pack_bf16x4(xv[ii]);
    }
    #pragma unroll
    for (int ii = 0; ii < 2; ++ii) {
      int i = tid + ii * 256;
      int r2 = (i >> 6) << 1;
      int c2 = (i & 63) << 1;
      *(unsigned int*)&sBt[b][c2][r2]     = pack_bf16x2(w0v[ii].x, w1v[ii].x);
      *(unsigned int*)&sBt[b][c2 + 1][r2] = pack_bf16x2(w0v[ii].y, w1v[ii].y);
    }
  };
  auto compute = [&](int b) {
    bf16x16 bfr[4];
    #pragma unroll
    for (int j = 0; j < 4; ++j)
      bfr[j] = load_frag_row(&sBt[b][64 * wc + 16 * j + nl][0], hi);
    #pragma unroll
    for (int i = 0; i < 2; ++i) {
      bf16x16 afr = load_frag_row(&sA[b][32 * wr + 16 * i + nl][0], hi);
      #pragma unroll
      for (int j = 0; j < 4; ++j) acc[i][j] = wmma_bf16(afr, bfr[j], acc[i][j]);
    }
  };

  // software pipeline
  load_regs(0);
  store_lds(0);
  __syncthreads();
  int buf = 0;
  for (int kt = 32; kt < K; kt += 32) {
    load_regs(kt);         // next tile's global loads in flight
    compute(buf);          // WMMA on current buffer hides load latency
    store_lds(buf ^ 1);    // wait loads, cvt, store to other buffer
    __syncthreads();
    buf ^= 1;
  }
  compute(buf);

  // C/D layout: vgpr r -> M = r + 8*hi, lane nl -> N
  #pragma unroll
  for (int i = 0; i < 2; ++i) {
    int mrow = m0 + 32 * wr + 16 * i + 8 * hi;
    #pragma unroll
    for (int j = 0; j < 4; ++j) {
      int ncol = n0 + 64 * wc + 16 * j + nl;
      #pragma unroll
      for (int r = 0; r < 8; ++r)
        C[(size_t)(mrow + r) * N + ncol] = acc[i][j][r];
    }
  }
}

// ---------------------------------------------------------------------------
// Kernel 2: split qkv, RMS-normalize q,k; write [H,M,D] layout to workspace
// ---------------------------------------------------------------------------
__global__ __launch_bounds__(128) void rms_split(const float* __restrict__ qkv,
                                                 float* __restrict__ qn,
                                                 float* __restrict__ kn,
                                                 float* __restrict__ vn,
                                                 int M, int H, int D, int N) {
  const int m = blockIdx.x >> 4;      // H == 16
  const int h = blockIdx.x & 15;
  const int d = threadIdx.x;
  const size_t base = (size_t)m * (3 * N) + (size_t)h * D;
  const float qv = qkv[base + d];
  const float kv = qkv[base + N + d];
  const float vv = qkv[base + 2 * N + d];

  __shared__ float red[2][4];
  float sq = qv * qv, sk = kv * kv;
  #pragma unroll
  for (int off = 16; off > 0; off >>= 1) {
    sq += __shfl_down(sq, off, 32);
    sk += __shfl_down(sk, off, 32);
  }
  if ((d & 31) == 0) { red[0][d >> 5] = sq; red[1][d >> 5] = sk; }
  __syncthreads();
  sq = red[0][0] + red[0][1] + red[0][2] + red[0][3];
  sk = red[1][0] + red[1][1] + red[1][2] + red[1][3];
  const float rq = rsqrtf(sq / (float)D);
  const float rk = rsqrtf(sk / (float)D);
  const size_t o = ((size_t)h * M + m) * D + d;
  qn[o] = qv * rq;
  kn[o] = kv * rk;
  vn[o] = vv;
}

// ---------------------------------------------------------------------------
// Kernel 3: attention over the virtually-updated cache.
// Grid: (M/32, H). 8 waves. L processed in 64-row chunks, double-buffered:
// next chunk's K/V global loads are in flight during both WMMA phases.
// Faithful to reference: plain exp, no max-subtraction, no 1/sqrt(D).
// ---------------------------------------------------------------------------
__global__ __launch_bounds__(256) void attn(const float* __restrict__ qn,
                                            const float* __restrict__ kn,
                                            const float* __restrict__ vn,
                                            const float* __restrict__ cK,
                                            const float* __restrict__ cV,
                                            float* __restrict__ out,
                                            const int* __restrict__ Pp,
                                            int M, int H, int L) {
  const int D = 128;
  const int P = *Pp;
  const int h  = blockIdx.y;
  const int m0 = blockIdx.x * 32;

  __shared__ __align__(16) unsigned short sQ[32][136];       // q tile   [m][d]
  __shared__ __align__(16) unsigned short sK[2][64][136];    // K chunk  [l][d]
  __shared__ __align__(16) unsigned short sVt[2][128][72];   // V chunk transposed [d][l]
  __shared__ __align__(16) unsigned short sP[32][72];        // exp(S)   [m][l]
  __shared__ float rowsum[32];

  const int tid = threadIdx.x, lane = tid & 31, wv = tid >> 5;
  const int nl = lane & 15, hi = lane >> 4;

  const float* qh  = qn + (size_t)h * M * D;
  const float* kh  = kn + (size_t)h * M * D;
  const float* vh  = vn + (size_t)h * M * D;
  const float* cKh = cK + (size_t)h * L * D;
  const float* cVh = cV + (size_t)h * L * D;

  {
    float4 qv4[4];
    #pragma unroll
    for (int ii = 0; ii < 4; ++ii) {
      int i = tid + ii * 256;               // 0..1023
      int r = i >> 5, d4 = (i & 31) << 2;
      qv4[ii] = *(const float4*)(qh + (size_t)(m0 + r) * D + d4);
    }
    #pragma unroll
    for (int ii = 0; ii < 4; ++ii) {
      int i = tid + ii * 256;
      int r = i >> 5, d4 = (i & 31) << 2;
      *(uint2*)&sQ[r][d4] = pack_bf16x4(qv4[ii]);
    }
  }
  if (tid < 32) rowsum[tid] = 0.0f;

  // out accumulators: wave -> mi = wv&1, col tiles (wv>>1) and (wv>>1)+4
  floatx8 oacc[2]; oacc[0] = (floatx8){}; oacc[1] = (floatx8){};
  const int mi = wv & 1;
  const int c0 = wv >> 1;

  float4 kv4[8];
  float2 va[8], vb[8];

  auto kv_load = [&](int l0) {
    #pragma unroll
    for (int ii = 0; ii < 8; ++ii) {
      int i = tid + ii * 256;               // 0..2047
      int r = i >> 5, d4 = (i & 31) << 2;
      int l = l0 + r;
      const float* krow = (l >= P && l < P + M) ? (kh + (size_t)(l - P) * D)
                                                : (cKh + (size_t)l * D);
      kv4[ii] = *(const float4*)(krow + d4);
    }
    #pragma unroll
    for (int ii = 0; ii < 8; ++ii) {
      int i = tid + ii * 256;               // 0..2047
      int d2 = (i & 63) << 1, l2 = (i >> 6) << 1;
      int la = l0 + l2, lb = la + 1;
      const float* vra = (la >= P && la < P + M) ? (vh + (size_t)(la - P) * D)
                                                 : (cVh + (size_t)la * D);
      const float* vrb = (lb >= P && lb < P + M) ? (vh + (size_t)(lb - P) * D)
                                                 : (cVh + (size_t)lb * D);
      va[ii] = *(const float2*)(vra + d2);
      vb[ii] = *(const float2*)(vrb + d2);
    }
  };
  auto kv_store = [&](int b) {
    #pragma unroll
    for (int ii = 0; ii < 8; ++ii) {
      int i = tid + ii * 256;
      int r = i >> 5, d4 = (i & 31) << 2;
      *(uint2*)&sK[b][r][d4] = pack_bf16x4(kv4[ii]);
    }
    #pragma unroll
    for (int ii = 0; ii < 8; ++ii) {
      int i = tid + ii * 256;
      int d2 = (i & 63) << 1, l2 = (i >> 6) << 1;
      *(unsigned int*)&sVt[b][d2][l2]     = pack_bf16x2(va[ii].x, vb[ii].x);
      *(unsigned int*)&sVt[b][d2 + 1][l2] = pack_bf16x2(va[ii].y, vb[ii].y);
    }
  };
  auto s_phase = [&](int b) {
    const int smi = wv & 1;            // S row tile
    const int sni = wv >> 1;           // S col tile (0..3)
    const int arow = 16 * smi + nl;    // q row for this lane
    const int bcol = 16 * sni + nl;    // cache row l for this lane
    floatx8 s = (floatx8){};
    #pragma unroll
    for (int d0 = 0; d0 < D; d0 += 32) {
      bf16x16 af  = load_frag_row(&sQ[arow][d0], hi);
      bf16x16 bfK = load_frag_row(&sK[b][bcol][d0], hi);   // B = K^T: row read
      s = wmma_bf16(af, bfK, s);
    }
    #pragma unroll
    for (int r = 0; r < 8; ++r) {
      float e = __expf(s[r]);
      int mloc = 16 * smi + 8 * hi + r;
      atomicAdd(&rowsum[mloc], e);                         // ds_add_f32
      sP[mloc][16 * sni + nl] = bf16_bits(e);
    }
  };
  auto pv_phase = [&](int b) {
    #pragma unroll
    for (int t = 0; t < 2; ++t) {
      const int ct   = c0 + 4 * t;       // out col tile 0..7
      const int vcol = 16 * ct + nl;     // d index
      const int prow = 16 * mi + nl;
      floatx8 o = oacc[t];
      #pragma unroll
      for (int k0 = 0; k0 < 64; k0 += 32) {
        bf16x16 af  = load_frag_row(&sP[prow][k0], hi);
        bf16x16 bfV = load_frag_row(&sVt[b][vcol][k0], hi); // B = V: row read
        o = wmma_bf16(af, bfV, o);
      }
      oacc[t] = o;
    }
  };

  // software pipeline over the 64-row L chunks
  kv_load(0);
  kv_store(0);
  __syncthreads();
  int buf = 0;
  for (int l0 = 0; l0 < L; l0 += 64) {
    const bool more = (l0 + 64) < L;
    if (more) kv_load(l0 + 64);   // next chunk's HBM traffic in flight
    s_phase(buf);                 // q@K^T + exp + rowsum + sP
    __syncthreads();              // sP ready for all waves
    pv_phase(buf);                // P@V accumulate
    if (more) kv_store(buf ^ 1);  // drain loads into the other buffer
    __syncthreads();              // next chunk staged / sP reusable
    buf ^= 1;
  }

  // normalize by softmax denominator and store [M, H*D]
  #pragma unroll
  for (int t = 0; t < 2; ++t) {
    const int ct = c0 + 4 * t;
    const int mloc = 16 * mi + 8 * hi;
    #pragma unroll
    for (int r = 0; r < 8; ++r) {
      float rs = rowsum[mloc + r];
      out[(size_t)(m0 + mloc + r) * (H * 128) + h * 128 + 16 * ct + nl] =
          oacc[t][r] / rs;
    }
  }
}

// ---------------------------------------------------------------------------
extern "C" void kernel_launch(void* const* d_in, const int* in_sizes, int n_in,
                              void* d_out, int out_size, void* d_ws, size_t ws_size,
                              hipStream_t stream) {
  (void)in_sizes; (void)n_in; (void)out_size; (void)ws_size;
  const float* X   = (const float*)d_in[0];   // [512, 2048]
  const float* W   = (const float*)d_in[1];   // [2048, 6144]
  const float* cK  = (const float*)d_in[2];   // [16, 4096, 128]
  const float* cV  = (const float*)d_in[3];   // [16, 4096, 128]
  const int*   Pp  = (const int*)d_in[4];     // scalar P

  const int M = 512, N = 2048, N3 = 3 * N, D = 128, H = 16, L = 4096;

  float* qkv = (float*)d_ws;                        // [512, 6144]
  float* qn  = qkv + (size_t)M * N3;                // [16, 512, 128]
  float* kn  = qn  + (size_t)H * M * D;
  float* vn  = kn  + (size_t)H * M * D;
  float* out = (float*)d_out;                       // [512, 2048]

  qkv_gemm<<<dim3(N3 / 128, M / 128), 256, 0, stream>>>(X, W, qkv, M, N3, N);
  rms_split<<<M * H, 128, 0, stream>>>(qkv, qn, kn, vn, M, H, D, N);
  attn<<<dim3(M / 32, H), 256, 0, stream>>>(qn, kn, vn, cK, cV, out, Pp, M, H, L);
}
